// GINBase_76124000354606
// MI455X (gfx1250) — matrix operand order repaired and checked by reading, not compiled, not run on hardware
//
#include <hip/hip_runtime.h>
#include <hip/hip_bf16.h>

#define NN 50000
#define EE 512000
#define DD 64
#define LL 4

typedef __attribute__((ext_vector_type(16))) __bf16 v16bf;
typedef __attribute__((ext_vector_type(8)))  __bf16 v8bf;
typedef __attribute__((ext_vector_type(2)))  __bf16 v2bf;
typedef __attribute__((ext_vector_type(8)))  float  v8f;

union ABFrag { v16bf v; v8bf h[2]; };

// A fragment: 16x32 bf16 tile, A row-major [16][ld] in LDS, column offset k0.
// ISA lane layout: lanes 0-15 hold K {k0..k0+7, k0+16..k0+23}, lanes 16-31 hold {+8,+24}.
__device__ __forceinline__ v16bf load_a_frag(const __bf16* base, int ld, int k0, int lane) {
    int row = lane & 15;
    int g   = (lane >> 4) << 3;            // 0 or 8
    const __bf16* p = base + row * ld + k0;
    ABFrag f;
    f.h[0] = *(const v8bf*)(p + g);
    f.h[1] = *(const v8bf*)(p + 16 + g);
    return f.v;
}

// B fragment: 32x16 (KxN) tile from transposed weights Wt[N][K] row-major in LDS.
__device__ __forceinline__ v16bf load_b_frag(const __bf16* wt, int ldk, int nbase, int k0, int lane) {
    int n = lane & 15;
    int g = (lane >> 4) << 3;              // 0 or 8
    const __bf16* p = wt + (size_t)(nbase + n) * ldk + k0;
    ABFrag f;
    f.h[0] = *(const v8bf*)(p + g);
    f.h[1] = *(const v8bf*)(p + 16 + g);
    return f.v;
}

__device__ __forceinline__ v8f wmma_bf16(v16bf a, v16bf b, v8f c) {
    return __builtin_amdgcn_wmma_f32_16x16x32_bf16(false, a, false, b, (short)0, c, false, false);
}

__device__ __forceinline__ v2bf pack_bf16(float a, float b) {
    v2bf r; r[0] = (__bf16)a; r[1] = (__bf16)b; return r;
}

// ---------------- encoders ----------------
__global__ __launch_bounds__(256) void k_encode_nodes(const int* __restrict__ x,
                                                      const float* __restrict__ atom_emb,
                                                      float* __restrict__ node) {
    int tid = blockIdx.x * 256 + threadIdx.x;
    if (tid >= NN * DD) return;
    int n = tid >> 6, d = tid & 63;
    float s = 0.f;
#pragma unroll
    for (int k = 0; k < 9; ++k) {
        int idx = x[n * 9 + k];
        s += atom_emb[((size_t)(k * 119 + idx)) * DD + d];
    }
    node[tid] = s;
}

__global__ __launch_bounds__(256) void k_encode_edges(const int* __restrict__ ea,
                                                      const float* __restrict__ bond_emb,
                                                      float* __restrict__ edgef) {
    int tid = blockIdx.x * 256 + threadIdx.x;
    if (tid >= EE * DD) return;
    int e = tid >> 6, d = tid & 63;
    float s = 0.f;
#pragma unroll
    for (int k = 0; k < 3; ++k) {
        int idx = ea[e * 3 + k];
        s += bond_emb[((size_t)(k * 6 + idx)) * DD + d];
    }
    edgef[tid] = s;
}

// cast f32 [L][R][C] -> bf16 transposed [L][C][R]
__global__ __launch_bounds__(256) void k_cast_transpose(const float* __restrict__ src,
                                                        __bf16* __restrict__ dst,
                                                        int Ln, int R, int C) {
    int total = Ln * R * C;
    for (int i = blockIdx.x * 256 + threadIdx.x; i < total; i += gridDim.x * 256) {
        int l = i / (R * C);
        int rc = i - l * R * C;
        int r = rc / C, c = rc - r * C;
        dst[(size_t)l * R * C + (size_t)c * R + r] = (__bf16)src[i];
    }
}

// ---------------- message + scatter-add (float4 per thread, fp32 atomics) ----------------
__global__ __launch_bounds__(256) void k_msg_scatter(const float* __restrict__ node,
                                                     const float* __restrict__ edgef,
                                                     const int* __restrict__ ei,
                                                     float* __restrict__ agg) {
    int tid = blockIdx.x * 256 + threadIdx.x;
    if (tid >= EE * 16) return;
    int e  = tid >> 4;
    int d0 = (tid & 15) << 2;
    int s = ei[e];
    int t = ei[EE + e];
    float4 nv = *(const float4*)(node + (size_t)s * DD + d0);
    float4 ev = *(const float4*)(edgef + (size_t)e * DD + d0);
    float* a = agg + (size_t)t * DD + d0;
    atomicAdd(a + 0, fmaxf(nv.x + ev.x, 0.f));
    atomicAdd(a + 1, fmaxf(nv.y + ev.y, 0.f));
    atomicAdd(a + 2, fmaxf(nv.z + ev.z, 0.f));
    atomicAdd(a + 3, fmaxf(nv.w + ev.w, 0.f));
}

// ---------------- node MLP: 32-row tiles, 8 balanced waves ----------------
__global__ __launch_bounds__(256) void k_node_mlp(float* __restrict__ node,
                                                  const float* __restrict__ agg,
                                                  const __bf16* __restrict__ W1t,
                                                  const __bf16* __restrict__ W2t,
                                                  const float* __restrict__ b1,
                                                  const float* __restrict__ b2,
                                                  const float* __restrict__ ln_g,
                                                  const float* __restrict__ ln_b,
                                                  const float* __restrict__ eps_arr,
                                                  int layer) {
    extern __shared__ __align__(32) char smem[];
    __bf16* sW1 = (__bf16*)smem;            // [128][64]  Nout x K
    __bf16* sW2 = sW1 + 128 * 64;           // [64][128]
    __bf16* sA  = sW2 + 64 * 128;           // [32][64]
    __bf16* sH1 = sA + 32 * 64;             // [32][128]
    float*  sH2 = (float*)(sH1 + 32 * 128); // [32][64]

    int tid = threadIdx.x, lane = tid & 31, wave = tid >> 5;
    for (int i = tid; i < 128 * 64; i += 256) sW1[i] = W1t[layer * 128 * 64 + i];
    for (int i = tid; i < 64 * 128; i += 256) sW2[i] = W2t[layer * 64 * 128 + i];
    float ep = 1.0f + eps_arr[layer];
    const float* b1l = b1 + layer * 128;
    const float* b2l = b2 + layer * 64;
    const float* gl  = ln_g + layer * 64;
    const float* bl  = ln_b + layer * 64;

    int numTiles = (NN + 31) / 32;  // 1563 (last tile: 16 valid rows)
    for (int t = blockIdx.x; t < numTiles; t += gridDim.x) {
        int r0 = t * 32;
        __syncthreads();
        // build A: 32x64 bf16 (pairs, guarded tail)
        for (int p = tid; p < 32 * 32; p += 256) {        // pair index (2 cols each)
            int i = p << 1;
            int row = i >> 6, d = i & 63;
            int n = r0 + row;
            v2bf pk;
            if (n < NN) {
                const float2 nv = *(const float2*)(node + (size_t)n * DD + d);
                const float2 av = *(const float2*)(agg  + (size_t)n * DD + d);
                pk = pack_bf16(ep * nv.x + av.x, ep * nv.y + av.y);
            } else {
                pk = pack_bf16(0.f, 0.f);
            }
            *(v2bf*)(sA + i) = pk;
        }
        __syncthreads();
        // GEMM1 [32x64]@[64x128], relu -> sH1 (bf16): 16 wave-tiles, 2 per wave
#pragma unroll
        for (int it = 0; it < 2; ++it) {
            int idx = wave + it * 8;         // 0..15
            int mt = idx >> 3, ct = idx & 7;
            v8f acc = {};
#pragma unroll
            for (int k0 = 0; k0 < 64; k0 += 32) {
                v16bf a = load_a_frag(sA + mt * 16 * 64, 64, k0, lane);
                v16bf b = load_b_frag(sW1, 64, ct * 16, k0, lane);
                acc = wmma_bf16(a, b, acc);
            }
            int c  = ct * 16 + (lane & 15);
            int mb = mt * 16 + ((lane >> 4) << 3);
            float bias = b1l[c];
#pragma unroll
            for (int r = 0; r < 8; ++r)
                sH1[(mb + r) * 128 + c] = (__bf16)fmaxf(acc[r] + bias, 0.f);
        }
        __syncthreads();
        // GEMM2 [32x128]@[128x64] + b2 -> sH2 (f32): 8 wave-tiles, 1 per wave
        {
            int mt = wave >> 2, ct = wave & 3;
            v8f acc = {};
#pragma unroll
            for (int k0 = 0; k0 < 128; k0 += 32) {
                v16bf a = load_a_frag(sH1 + mt * 16 * 128, 128, k0, lane);
                v16bf b = load_b_frag(sW2, 128, ct * 16, k0, lane);
                acc = wmma_bf16(a, b, acc);
            }
            int c  = ct * 16 + (lane & 15);
            int mb = mt * 16 + ((lane >> 4) << 3);
            float bias = b2l[c];
#pragma unroll
            for (int r = 0; r < 8; ++r) sH2[(mb + r) * 64 + c] = acc[r] + bias;
        }
        __syncthreads();
        // LayerNorm(64) + relu + residual -> node  (32 rows x 8 threads, 8 cols each)
        {
            int row = tid >> 3, j = tid & 7;
            float s = 0.f, ss = 0.f;
#pragma unroll
            for (int q = 0; q < 8; ++q) {
                float v = sH2[row * 64 + j * 8 + q];
                s += v; ss += v * v;
            }
#pragma unroll
            for (int o = 4; o >= 1; o >>= 1) {
                s  += __shfl_xor(s, o, 8);
                ss += __shfl_xor(ss, o, 8);
            }
            float mean = s * (1.f / 64.f);
            float var  = ss * (1.f / 64.f) - mean * mean;
            float inv  = rsqrtf(var + 1e-5f);
            int n = r0 + row;
            if (n < NN) {
                size_t nb = (size_t)n * DD;
#pragma unroll
                for (int q = 0; q < 8; ++q) {
                    int c = j * 8 + q;
                    float y = (sH2[row * 64 + c] - mean) * inv * gl[c] + bl[c];
                    node[nb + c] = fmaxf(y, 0.f) + node[nb + c];
                }
            }
        }
    }
}

// ---------------- edge update: 32-edge tiles, 8 balanced waves ----------------
__global__ __launch_bounds__(256) void k_edge_update(const float* __restrict__ node,
                                                     float* __restrict__ edgef,
                                                     const int* __restrict__ ei,
                                                     const __bf16* __restrict__ Wu1t,
                                                     const __bf16* __restrict__ Wu2t,
                                                     const float* __restrict__ bu1,
                                                     const float* __restrict__ bu2,
                                                     const float* __restrict__ ug,
                                                     const float* __restrict__ ub,
                                                     int layer) {
    extern __shared__ __align__(32) char smem[];
    __bf16* sW1  = (__bf16*)smem;             // [192][192] Nout x K
    __bf16* sW2  = sW1 + 192 * 192;           // [64][192]
    __bf16* sCat = sW2 + 64 * 192;            // [32][192]  A matrix (cat, then relu(LN(m)))
    float*  sM   = (float*)(sCat + 32 * 192); // [32][192] (also reused as [32][64] out stage)

    int tid = threadIdx.x, lane = tid & 31, wave = tid >> 5;
    for (int i = tid; i < 192 * 192; i += 256) sW1[i] = Wu1t[layer * 192 * 192 + i];
    for (int i = tid; i < 64 * 192; i += 256)  sW2[i] = Wu2t[layer * 64 * 192 + i];
    const float* bu1l = bu1 + layer * 192;
    const float* bu2l = bu2 + layer * 64;
    const float* gl   = ug + layer * 192;
    const float* bl   = ub + layer * 192;

    int numTiles = EE / 32;  // 16000
    for (int t = blockIdx.x; t < numTiles; t += gridDim.x) {
        int e0 = t * 32;
        // prefetch next tile's indices while we compute this one
        int tn = t + gridDim.x;
        if (tn < numTiles) {
            __builtin_prefetch(&ei[tn * 32], 0, 0);
            __builtin_prefetch(&ei[EE + tn * 32], 0, 0);
        }
        __syncthreads();
        // build cat = [node[src], node[dst], edge] as bf16 pairs: 32x192
        for (int p = tid; p < 32 * 96; p += 256) {
            int i = p << 1;
            int row = i / 192, c = i - row * 192;
            int e = e0 + row;
            float2 v;
            if (c < 64)       v = *(const float2*)(node + (size_t)ei[e] * DD + c);
            else if (c < 128) v = *(const float2*)(node + (size_t)ei[EE + e] * DD + (c - 64));
            else              v = *(const float2*)(edgef + (size_t)e * DD + (c - 128));
            *(v2bf*)(sCat + i) = pack_bf16(v.x, v.y);
        }
        __syncthreads();
        // GEMM1 [32x192]@[192x192] + bu1 -> sM: 24 wave-tiles, 3 per wave
#pragma unroll
        for (int it = 0; it < 3; ++it) {
            int idx = wave + it * 8;        // 0..23
            int mt = idx / 12, ct = idx - mt * 12;
            v8f acc = {};
#pragma unroll
            for (int k0 = 0; k0 < 192; k0 += 32) {
                v16bf a = load_a_frag(sCat + mt * 16 * 192, 192, k0, lane);
                v16bf b = load_b_frag(sW1, 192, ct * 16, k0, lane);
                acc = wmma_bf16(a, b, acc);
            }
            int c  = ct * 16 + (lane & 15);
            int mb = mt * 16 + ((lane >> 4) << 3);
            float bias = bu1l[c];
#pragma unroll
            for (int r = 0; r < 8; ++r) sM[(mb + r) * 192 + c] = acc[r] + bias;
        }
        __syncthreads();
        // LayerNorm(192) + relu -> sCat (bf16): 32 rows x 8 threads, 24 cols each
        {
            int row = tid >> 3, j = tid & 7;
            float s = 0.f, ss = 0.f;
#pragma unroll
            for (int q = 0; q < 24; ++q) {
                float v = sM[row * 192 + j * 24 + q];
                s += v; ss += v * v;
            }
#pragma unroll
            for (int o = 4; o >= 1; o >>= 1) {
                s  += __shfl_xor(s, o, 8);
                ss += __shfl_xor(ss, o, 8);
            }
            float mean = s * (1.f / 192.f);
            float var  = ss * (1.f / 192.f) - mean * mean;
            float inv  = rsqrtf(var + 1e-5f);
#pragma unroll
            for (int q = 0; q < 24; q += 2) {
                int c = j * 24 + q;
                float y0 = (sM[row * 192 + c]     - mean) * inv * gl[c]     + bl[c];
                float y1 = (sM[row * 192 + c + 1] - mean) * inv * gl[c + 1] + bl[c + 1];
                *(v2bf*)(sCat + row * 192 + c) = pack_bf16(fmaxf(y0, 0.f), fmaxf(y1, 0.f));
            }
        }
        __syncthreads();
        // GEMM2 [32x192]@[192x64] + bu2 -> sM[32][64]: 8 wave-tiles, 1 per wave
        {
            int mt = wave >> 2, ct = wave & 3;
            v8f acc = {};
#pragma unroll
            for (int k0 = 0; k0 < 192; k0 += 32) {
                v16bf a = load_a_frag(sCat + mt * 16 * 192, 192, k0, lane);
                v16bf b = load_b_frag(sW2, 192, ct * 16, k0, lane);
                acc = wmma_bf16(a, b, acc);
            }
            int c  = ct * 16 + (lane & 15);
            int mb = mt * 16 + ((lane >> 4) << 3);
            float bias = bu2l[c];
#pragma unroll
            for (int r = 0; r < 8; ++r) sM[(mb + r) * 64 + c] = acc[r] + bias;
        }
        __syncthreads();
        // cooperative coalesced residual + store: 32x64 f32, float4 per thread x2
#pragma unroll
        for (int q = 0; q < 2; ++q) {
            int i4 = (tid + q * 256) << 2;          // float index, 16B aligned
            int row = i4 >> 6, c = i4 & 63;
            size_t off = (size_t)(e0 + row) * DD + c;
            float4 res = *(const float4*)(edgef + off);
            float4 v   = *(const float4*)(sM + row * 64 + c);
            v.x += res.x; v.y += res.y; v.z += res.z; v.w += res.w;
            *(float4*)(edgef + off) = v;
        }
    }
}

extern "C" void kernel_launch(void* const* d_in, const int* in_sizes, int n_in,
                              void* d_out, int out_size, void* d_ws, size_t ws_size,
                              hipStream_t stream) {
    const int*   x         = (const int*)d_in[0];
    const int*   edge_attr = (const int*)d_in[1];
    const int*   ei        = (const int*)d_in[2];
    const float* atom_emb  = (const float*)d_in[3];
    const float* bond_emb  = (const float*)d_in[4];
    const float* eps       = (const float*)d_in[5];
    const float* W1        = (const float*)d_in[6];
    const float* b1        = (const float*)d_in[7];
    const float* W2        = (const float*)d_in[8];
    const float* b2        = (const float*)d_in[9];
    const float* ln_g      = (const float*)d_in[10];
    const float* ln_b      = (const float*)d_in[11];
    const float* Wu1       = (const float*)d_in[12];
    const float* bu1       = (const float*)d_in[13];
    const float* ug        = (const float*)d_in[14];
    const float* ub        = (const float*)d_in[15];
    const float* Wu2       = (const float*)d_in[16];
    const float* bu2       = (const float*)d_in[17];

    float* node  = (float*)d_out;                 // [N,64]
    float* edgef = node + (size_t)NN * DD;        // [E,64]

    float*  agg  = (float*)d_ws;                  // [N,64] f32
    size_t aggB  = (((size_t)NN * DD * sizeof(float)) + 255) & ~(size_t)255;
    __bf16* W1t  = (__bf16*)((char*)d_ws + aggB); // [L][128][64]
    __bf16* W2t  = W1t + LL * 128 * 64;           // [L][64][128]
    __bf16* Wu1t = W2t + LL * 64 * 128;           // [L][192][192]
    __bf16* Wu2t = Wu1t + LL * 192 * 192;         // [L][64][192]

    k_encode_nodes<<<(NN * DD + 255) / 256, 256, 0, stream>>>(x, atom_emb, node);
    k_encode_edges<<<(EE * DD + 255) / 256, 256, 0, stream>>>(edge_attr, bond_emb, edgef);

    k_cast_transpose<<<(LL * 64 * 128 + 255) / 256, 256, 0, stream>>>(W1, W1t, LL, 64, 128);
    k_cast_transpose<<<(LL * 128 * 64 + 255) / 256, 256, 0, stream>>>(W2, W2t, LL, 128, 64);
    k_cast_transpose<<<(LL * 192 * 192 + 255) / 256, 256, 0, stream>>>(Wu1, Wu1t, LL, 192, 192);
    k_cast_transpose<<<(LL * 192 * 64 + 255) / 256, 256, 0, stream>>>(Wu2, Wu2t, LL, 192, 64);

    const size_t nodeSmem = (128 * 64 + 64 * 128 + 32 * 64 + 32 * 128) * sizeof(__bf16)
                          + 32 * 64 * sizeof(float);                         // ~52 KB
    const size_t edgeSmem = (192 * 192 + 64 * 192 + 32 * 192) * sizeof(__bf16)
                          + 32 * 192 * sizeof(float);                        // ~132 KB

    for (int l = 0; l < LL; ++l) {
        hipMemsetAsync(agg, 0, (size_t)NN * DD * sizeof(float), stream);
        k_msg_scatter<<<(EE * 16 + 255) / 256, 256, 0, stream>>>(node, edgef, ei, agg);
        k_node_mlp<<<512, 256, nodeSmem, stream>>>(node, agg, W1t, W2t, b1, b2,
                                                   ln_g, ln_b, eps, l);
        k_edge_update<<<1024, 256, edgeSmem, stream>>>(node, edgef, ei, Wu1t, Wu2t,
                                                       bu1, bu2, ug, ub, l);
    }
}
